// BatchESN_29532195127499
// MI455X (gfx1250) — matrix operand order repaired and checked by reading, not compiled, not run on hardware
//
#include <hip/hip_runtime.h>
#include <hip/hip_fp16.h>
#include <math.h>

typedef __attribute__((ext_vector_type(16))) _Float16 v16h;
typedef __attribute__((ext_vector_type(8)))  _Float16 v8h;
typedef __attribute__((ext_vector_type(8)))  float    v8f;

#define RDIM 4096
#define IDIM 64
#define ODIM 64
#define TDIM 8192
#define CR   4160          // IDIM + RDIM (concat row length), = 130 * 32
#define KT_RES 128         // RDIM / 32
#define KT_OUT 130         // CR / 32
#define NWG  64            // persistent workgroups for recurrence
#define NTHR 128           // 4 waves per WG, wave32

// ---------------------------------------------------------------------------
// barrier init
__global__ void zero_bar_k(unsigned* b) { b[0] = 0u; b[1] = 0u; }

// ---------------------------------------------------------------------------
// Pack W_res (R x R f32 row-major) into WMMA A-fragment order, f16.
// Tile (rb, kt): lane holds row rb*16+(lane&15); elems e: K = (lane>=16?8:0)+(e<8?e:e+8)
__global__ void pack_wres_k(const float* __restrict__ W, _Float16* __restrict__ P) {
  int tid  = blockIdx.x * blockDim.x + threadIdx.x;   // 256*128*32 threads
  int lane = tid & 31;
  int tile = tid >> 5;                                 // rb*128 + kt
  int rb   = tile >> 7;
  int kt   = tile & 127;
  int r    = rb * 16 + (lane & 15);
  int cb   = kt * 32 + (lane >> 4) * 8;
  const float* src = W + (size_t)r * RDIM + cb;
  v16h a;
#pragma unroll
  for (int e = 0; e < 8; ++e) {
    a[e]     = (_Float16)src[e];
    a[e + 8] = (_Float16)src[16 + e];
  }
  *(v16h*)(P + (size_t)tid * 16) = a;
}

// ---------------------------------------------------------------------------
// Pack W_out (O x CR f32 row-major) into WMMA B-fragment order, f16.
// Tile (ob, kt): lane holds col ob*16+(lane&15); elems e: K = (lane>=16?8:0)+(e<8?e:e+8)
__global__ void pack_wout_k(const float* __restrict__ W, _Float16* __restrict__ P) {
  int tid = blockIdx.x * blockDim.x + threadIdx.x;
  if (tid >= 4 * KT_OUT * 32) return;
  int lane = tid & 31;
  int tile = tid >> 5;                                 // ob*130 + kt
  int ob   = tile / KT_OUT;
  int kt   = tile % KT_OUT;
  int o    = ob * 16 + (lane & 15);
  int kb   = kt * 32 + (lane >> 4) * 8;
  const float* src = W + (size_t)o * CR + kb;
  v16h b;
#pragma unroll
  for (int e = 0; e < 16; ++e) b[e] = (_Float16)src[e < 8 ? e : e + 8];
  *(v16h*)(P + (size_t)tid * 16) = b;
}

// ---------------------------------------------------------------------------
// Write f16 copy of X into the first IDIM columns of the concat buffer CS.
__global__ void pack_x_k(const float* __restrict__ X, _Float16* __restrict__ CS) {
  int tid = blockIdx.x * blockDim.x + threadIdx.x;     // TDIM*IDIM threads
  int t = tid >> 6;
  int i = tid & 63;
  CS[(size_t)t * CR + i] = (_Float16)X[tid];
}

// ---------------------------------------------------------------------------
// U[t][r] = sum_i W_in[r][i] * X[t][i]   (f32, precomputed input projection)
// block = (t-chunk of 16, r-chunk of 256)
__global__ void compute_u_k(const float* __restrict__ X, const float* __restrict__ Win,
                            float* __restrict__ U) {
  __shared__ float xs[16 * IDIM];
  int t0 = blockIdx.x * 16;
  int r  = blockIdx.y * 256 + threadIdx.x;
  for (int i = threadIdx.x; i < 16 * IDIM; i += 256) xs[i] = X[(size_t)t0 * IDIM + i];
  __syncthreads();
  float acc[16];
#pragma unroll
  for (int tt = 0; tt < 16; ++tt) acc[tt] = 0.f;
  const float* wr = Win + (size_t)r * IDIM;
  for (int i = 0; i < IDIM; ++i) {
    float w = wr[i];
#pragma unroll
    for (int tt = 0; tt < 16; ++tt) acc[tt] += w * xs[tt * IDIM + i];
  }
#pragma unroll
  for (int tt = 0; tt < 16; ++tt) U[(size_t)(t0 + tt) * RDIM + r] = acc[tt];
}

// ---------------------------------------------------------------------------
// Persistent recurrence: state_t = tanh(U_t + W_res @ state_{t-1}), grid barrier per step.
__global__ void __launch_bounds__(NTHR, 1)
recurrence_k(const float* __restrict__ U, const _Float16* __restrict__ WP,
             _Float16* __restrict__ CS, unsigned* __restrict__ bcount,
             unsigned* __restrict__ bgen) {
  __shared__ _Float16 sstate[RDIM];                    // 8 KB prev-state staging
  const int tid  = threadIdx.x;
  const int lane = tid & 31;
  const int wave = tid >> 5;
  const int rb   = (int)blockIdx.x * 4 + wave;         // 0..255, 16 rows each
  const _Float16* wbase = WP + (size_t)rb * KT_RES * 512 + (size_t)lane * 16;

  for (int t = 0; t < TDIM; ++t) {
    // stage previous state (zeros at t==0) into LDS
    float4* dst = (float4*)sstate;
    if (t == 0) {
      float4 z; z.x = 0.f; z.y = 0.f; z.z = 0.f; z.w = 0.f;
      for (int i = tid; i < RDIM / 8; i += NTHR) dst[i] = z;
    } else {
      const float4* src = (const float4*)(CS + (size_t)(t - 1) * CR + IDIM);
      for (int i = tid; i < RDIM / 8; i += NTHR) dst[i] = src[i];
    }
    __syncthreads();

    // 16-row GEMV chunk via WMMA: state replicated across all 16 B columns
    v8f acc = {};
    const _Float16* ap = wbase;
    const _Float16* bp = sstate + (lane >> 4) * 8;
#pragma unroll 4
    for (int kt = 0; kt < KT_RES; ++kt) {
      v16h a = *(const v16h*)ap;                       // coalesced, pre-packed
      v16h b = __builtin_shufflevector(*(const v8h*)bp, *(const v8h*)(bp + 16), 0, 1, 2, 3, 4, 5, 6, 7, 8, 9, 10, 11, 12, 13, 14, 15);                       // LDS broadcast
      acc = __builtin_amdgcn_wmma_f32_16x16x32_f16(false, a, false, b,
                                                   (short)0, acc, false, false);
      ap += 512;
      bp += 32;
    }

    // epilogue: column 0 of D == GEMV result; lanes 0 (rows 0-7) & 16 (rows 8-15)
    if ((lane & 15) == 0) {
      int r0 = rb * 16 + (lane >> 4) * 8;
#pragma unroll
      for (int e = 0; e < 8; ++e) {
        float v = tanhf(acc[e] + U[(size_t)t * RDIM + r0 + e]);
        CS[(size_t)t * CR + IDIM + r0 + e] = (_Float16)v;
      }
    }

    // device-wide sense/generation barrier
    __threadfence();
    __syncthreads();
    if (tid == 0) {
      unsigned prev = __hip_atomic_fetch_add(bcount, 1u, __ATOMIC_ACQ_REL,
                                             __HIP_MEMORY_SCOPE_AGENT);
      if (prev == NWG - 1) {
        __hip_atomic_store(bcount, 0u, __ATOMIC_RELAXED, __HIP_MEMORY_SCOPE_AGENT);
        __hip_atomic_fetch_add(bgen, 1u, __ATOMIC_RELEASE, __HIP_MEMORY_SCOPE_AGENT);
      } else {
        while (__hip_atomic_load(bgen, __ATOMIC_ACQUIRE, __HIP_MEMORY_SCOPE_AGENT)
               < (unsigned)(t + 1)) {
          __builtin_amdgcn_s_sleep(2);
        }
      }
    }
    __syncthreads();
    __threadfence();
  }
}

// ---------------------------------------------------------------------------
// Readout GEMM: Y[t][o] = CS[t] . W_out[o], WMMA over 130 K-tiles.
// block = t-tile (16 timesteps), wave = o-tile (16 outputs)
__global__ void __launch_bounds__(128)
readout_k(const _Float16* __restrict__ CS, const _Float16* __restrict__ WP,
          float* __restrict__ out) {
  int tid  = threadIdx.x;
  int lane = tid & 31;
  int ob   = tid >> 5;                                 // 0..3
  int t0   = (int)blockIdx.x * 16;
  const _Float16* arow  = CS + (size_t)(t0 + (lane & 15)) * CR + (lane >> 4) * 8;
  const _Float16* bbase = WP + ((size_t)ob * KT_OUT * 32 + lane) * 16;
  v8f acc = {};
#pragma unroll 2
  for (int kt = 0; kt < KT_OUT; ++kt) {
    v8h lo = *(const v8h*)arow;
    v8h hi = *(const v8h*)(arow + 16);
    v16h a = __builtin_shufflevector(lo, hi, 0, 1, 2, 3, 4, 5, 6, 7,
                                             8, 9, 10, 11, 12, 13, 14, 15);
    v16h b = *(const v16h*)(bbase + (size_t)kt * 512);
    acc = __builtin_amdgcn_wmma_f32_16x16x32_f16(false, a, false, b,
                                                 (short)0, acc, false, false);
    arow += 32;
  }
  int n  = lane & 15;
  int m0 = (lane >> 4) * 8;
#pragma unroll
  for (int e = 0; e < 8; ++e)
    out[(size_t)(t0 + m0 + e) * ODIM + ob * 16 + n] = acc[e];
}

// ---------------------------------------------------------------------------
extern "C" void kernel_launch(void* const* d_in, const int* in_sizes, int n_in,
                              void* d_out, int out_size, void* d_ws, size_t ws_size,
                              hipStream_t stream) {
  (void)in_sizes; (void)n_in; (void)out_size; (void)ws_size;
  const float* X    = (const float*)d_in[0];   // [T, I]
  const float* Win  = (const float*)d_in[1];   // [R, I]
  const float* Wres = (const float*)d_in[2];   // [R, R]
  const float* Wout = (const float*)d_in[3];   // [O, I+R]
  float* out = (float*)d_out;                  // [T, O]

  char* ws = (char*)d_ws;
  size_t off = 0;
  auto take = [&](size_t bytes) -> char* {
    char* p = ws + off;
    off += (bytes + 255) & ~(size_t)255;
    return p;
  };
  unsigned*  bar = (unsigned*)take(256);                           // [count, gen]
  _Float16*  CS  = (_Float16*)take((size_t)TDIM * CR * 2);         // concat [x_t | s_t] f16
  float*     U   = (float*)take((size_t)TDIM * RDIM * 4);         // input projection f32
  _Float16*  WrP = (_Float16*)take((size_t)RDIM * RDIM * 2);      // packed W_res f16
  _Float16*  WoP = (_Float16*)take((size_t)4 * KT_OUT * 32 * 16 * 2); // packed W_out f16

  zero_bar_k<<<1, 1, 0, stream>>>(bar);
  pack_wres_k<<<(256 * KT_RES * 32) / 256, 256, 0, stream>>>(Wres, WrP);
  pack_wout_k<<<(4 * KT_OUT * 32 + 255) / 256, 256, 0, stream>>>(Wout, WoP);
  pack_x_k<<<(TDIM * IDIM) / 256, 256, 0, stream>>>(X, CS);
  dim3 gu(TDIM / 16, RDIM / 256);
  compute_u_k<<<gu, 256, 0, stream>>>(X, Win, U);
  recurrence_k<<<NWG, NTHR, 0, stream>>>(U, WrP, CS, bar, bar + 1);
  readout_k<<<TDIM / 16, 128, 0, stream>>>(CS, WoP, out);
}
